// lnMatmulBlock_70824010711778
// MI455X (gfx1250) — matrix operand order repaired and checked by reading, not compiled, not run on hardware
//
#include <hip/hip_runtime.h>
#include <hip/hip_bf16.h>

// ---------------- CDNA5 types ----------------
typedef __attribute__((ext_vector_type(16))) __bf16 v16bf;
typedef __attribute__((ext_vector_type(8)))  float  v8f;

#define LN_EPS 1e-5f
#define BATCH 8
#define MDIM  4096
#define NDIM  1024   // LN / reduction (K) dim
#define DDIM  1024   // output dim
#define NROWS (BATCH * MDIM)   // 32768

#define BM 128   // M tile per workgroup
#define BD 128   // D tile per workgroup
#define BK 32    // K chunk (bf16 WMMA K)

// ============================================================
// Kernel 1: per-row LayerNorm statistics (mean, rstd)
// one 256-thread block per row of 1024 floats
// ============================================================
__global__ __launch_bounds__(256) void ln_stats_kernel(
    const float* __restrict__ x,
    float* __restrict__ mean_out,
    float* __restrict__ rstd_out) {
  const int row = blockIdx.x;
  const int t = threadIdx.x;

  const float4 v = ((const float4*)(x + (size_t)row * NDIM))[t];
  float s  = v.x + v.y + v.z + v.w;
  float ss = v.x * v.x + v.y * v.y + v.z * v.z + v.w * v.w;

  __shared__ float redS[256];
  __shared__ float redQ[256];
  redS[t] = s;
  redQ[t] = ss;
  __syncthreads();
  for (int off = 128; off > 0; off >>= 1) {
    if (t < off) {
      redS[t] += redS[t + off];
      redQ[t] += redQ[t + off];
    }
    __syncthreads();
  }
  if (t == 0) {
    const float m   = redS[0] * (1.0f / NDIM);
    const float var = redQ[0] * (1.0f / NDIM) - m * m;
    mean_out[row] = m;
    rstd_out[row] = rsqrtf(var + LN_EPS);
  }
}

// ============================================================
// Kernel 2: one-time weight conversion f32 -> bf16 (2 MB, L2-resident)
// ============================================================
__global__ __launch_bounds__(256) void w_to_bf16_kernel(
    const float* __restrict__ w, __bf16* __restrict__ wb) {
  const int i = (blockIdx.x * 256 + threadIdx.x) * 4;
  const float4 v = *(const float4*)(w + i);
  __bf16 o[4] = {(__bf16)v.x, (__bf16)v.y, (__bf16)v.z, (__bf16)v.w};
  *(uint2*)(wb + i) = *(const uint2*)o;
}

// ============================================================
// Kernel 3: fused (normalize -> bf16) x Wbf16 + bias via WMMA
// grid: (D/BD, NROWS/BM); block: 256 = 8 waves as 4(M) x 2(D)
// wave: 32 rows x 64 cols => 2 A frags, 4 B frags, 8 WMMA / K-chunk
// LDS double-buffered; global loads for chunk i+1 issued before
// compute on chunk i; single barrier per iteration.
// ============================================================
__global__ __launch_bounds__(256) void ln_wmma_gemm_kernel(
    const float* __restrict__ x,
    const __bf16* __restrict__ wb,
    const float* __restrict__ bias,
    const float* __restrict__ gamma,
    const float* __restrict__ beta,
    const float* __restrict__ mean,
    const float* __restrict__ rstd,
    float* __restrict__ out) {

  const int d0 = blockIdx.x * BD;            // D-tile origin
  const int mt = blockIdx.y;                 // flat M-tile 0..255
  const size_t rowbase = (size_t)mt * BM;    // global row (b*4096+m) of tile

  __shared__ __align__(32) __bf16 Als[2][BM][BK];   // A, row-major [m][k]
  __shared__ __align__(32) __bf16 Bls[2][BD][BK];   // B, TRANSPOSED [d][k]
  __shared__ float gLs[NDIM];
  __shared__ float bLs[NDIM];

  const int t    = threadIdx.x;
  const int lane = t & 31;
  const int wave = t >> 5;
  const int wm   = wave & 3;   // M strip: rows [wm*32, wm*32+32)
  const int wd   = wave >> 2;  // D half:  cols [wd*64, wd*64+64)

  // cache gamma/beta in LDS (1024 floats each, 4 per thread)
  {
    ((float4*)gLs)[t] = ((const float4*)gamma)[t];
    ((float4*)bLs)[t] = ((const float4*)beta)[t];
  }

  // ---- accumulators: 2(M) x 4(D) tiles of 16x16 f32 ----
  v8f acc[2][4];
#pragma unroll
  for (int mi = 0; mi < 2; ++mi)
#pragma unroll
    for (int di = 0; di < 4; ++di)
#pragma unroll
      for (int j = 0; j < 8; ++j) acc[mi][di][j] = 0.0f;

  // ---- A staging: thread t loads 16 floats of row t/2 ----
  const int ar = t >> 1;            // 0..127
  const int ak = (t & 1) * 16;      // 0 or 16
  const float mu = mean[rowbase + ar];
  const float rs = rstd[rowbase + ar];
  const float* xrow = x + (rowbase + ar) * (size_t)NDIM + ak;

  // ---- B staging: thread t loads 16 consecutive d (bf16), fixed k ----
  const int bd = (t * 16) & (BD - 1);   // 0..112 step 16
  const int bk = (t * 16) >> 7;         // 0..31
  const __bf16* wrow = wb + (size_t)bk * DDIM + d0 + bd;

  // register staging for next chunk
  float4 rA[4];
  uint4  rB[2];

  // ---- prologue: fetch chunk 0 ----
#pragma unroll
  for (int i = 0; i < 4; ++i) rA[i] = ((const float4*)xrow)[i];
  rB[0] = ((const uint4*)wrow)[0];
  rB[1] = ((const uint4*)wrow)[1];

  auto stage = [&](int buf, int k0, const float4* a4, const uint4* b4) {
#pragma unroll
    for (int i = 0; i < 4; ++i) {
      const float4 v = a4[i];
      const int kk = ak + i * 4;
      Als[buf][ar][kk + 0] = (__bf16)((v.x - mu) * rs * gLs[k0 + kk + 0] + bLs[k0 + kk + 0]);
      Als[buf][ar][kk + 1] = (__bf16)((v.y - mu) * rs * gLs[k0 + kk + 1] + bLs[k0 + kk + 1]);
      Als[buf][ar][kk + 2] = (__bf16)((v.z - mu) * rs * gLs[k0 + kk + 2] + bLs[k0 + kk + 2]);
      Als[buf][ar][kk + 3] = (__bf16)((v.w - mu) * rs * gLs[k0 + kk + 3] + bLs[k0 + kk + 3]);
    }
    const __bf16* bsrc = (const __bf16*)b4;
#pragma unroll
    for (int i = 0; i < 16; ++i) Bls[buf][bd + i][bk] = bsrc[i];
  };

  __syncthreads();          // gLs/bLs ready
  stage(0, 0, rA, rB);
  __syncthreads();          // buffer 0 ready

  for (int k0 = 0; k0 < NDIM; k0 += BK) {
    const int cur = (k0 >> 5) & 1;
    const int nxt = cur ^ 1;
    const bool have_next = (k0 + BK) < NDIM;

    // ---- issue global loads for next chunk first (latency hiding) ----
    if (have_next) {
      const float4* xp = (const float4*)(xrow + k0 + BK);
#pragma unroll
      for (int i = 0; i < 4; ++i) rA[i] = xp[i];
      const uint4* wp = (const uint4*)(wrow + (size_t)BK * DDIM +
                                       ((size_t)(k0 >> 5)) * BK * DDIM);
      rB[0] = wp[0];
      rB[1] = wp[1];
    }
    if (k0 + 2 * BK < NDIM) {
      __builtin_prefetch(xrow + k0 + 2 * BK, 0, 1);     // global_prefetch_b8
    }

    // ---- compute from current buffer ----
    // A frag (16x32 bf16): lanes 0-15: M=lane, K=0..15;
    //                      lanes 16-31: M=lane-16, K=16..31
    const int kb = (lane >> 4) * 16;
    v16bf afrag[2];
#pragma unroll
    for (int mi = 0; mi < 2; ++mi) {
      const int arow = wm * 32 + mi * 16 + (lane & 15);
      uint4* ap = (uint4*)&afrag[mi];
      ap[0] = *(const uint4*)&Als[cur][arow][kb];
      ap[1] = *(const uint4*)&Als[cur][arow][kb + 8];
    }
    // B frag (32x16 bf16): lanes 0-15 hold K=0..15 of col N=lane;
    // lanes 16-31 hold K=16..31 -> contiguous 32B in transposed tile.
    const int bkb = (lane >> 4) * 16;
#pragma unroll
    for (int di = 0; di < 4; ++di) {
      const int bcol = wd * 64 + di * 16 + (lane & 15);
      const v16bf bfrag = *(const v16bf*)&Bls[cur][bcol][bkb];
#pragma unroll
      for (int mi = 0; mi < 2; ++mi) {
        acc[mi][di] = __builtin_amdgcn_wmma_f32_16x16x32_bf16(
            /*neg_a=*/false, afrag[mi], /*neg_b=*/false, bfrag,
            /*c_mod=*/(short)0, acc[mi][di],
            /*reuse_a=*/false, /*reuse_b=*/false);
      }
    }

    // ---- store next chunk into the other buffer ----
    if (have_next) stage(nxt, k0 + BK, rA, rB);
    __syncthreads();
  }

  // ---- epilogue: C/D layout (VGPR j: M=j lanes 0-15, M=j+8 lanes 16-31) ----
  const int crow0 = wm * 32 + (lane >> 4) * 8;
  const int ccol  = lane & 15;
#pragma unroll
  for (int di = 0; di < 4; ++di) {
    const int dcol = d0 + wd * 64 + di * 16 + ccol;
    const float bb = bias[dcol];
#pragma unroll
    for (int mi = 0; mi < 2; ++mi) {
#pragma unroll
      for (int j = 0; j < 8; ++j) {
        const size_t orow = rowbase + crow0 + mi * 16 + j;
        out[orow * (size_t)DDIM + dcol] = acc[mi][di][j] + bb;
      }
    }
  }
}

// ============================================================
// launcher
// ============================================================
extern "C" void kernel_launch(void* const* d_in, const int* in_sizes, int n_in,
                              void* d_out, int out_size, void* d_ws, size_t ws_size,
                              hipStream_t stream) {
  const float* x     = (const float*)d_in[0];   // [8,4096,1024]
  const float* w     = (const float*)d_in[1];   // [1024,1024]
  const float* bias  = (const float*)d_in[2];   // [1024]
  const float* gamma = (const float*)d_in[3];   // [1024]
  const float* beta  = (const float*)d_in[4];   // [1024]
  float* out = (float*)d_out;                   // [8,4096,1024]

  float*  mean = (float*)d_ws;                  // 32768 f32
  float*  rstd = mean + NROWS;                  // 32768 f32
  __bf16* wbf  = (__bf16*)(rstd + NROWS);       // 1M bf16 (2 MB)

  ln_stats_kernel<<<NROWS, 256, 0, stream>>>(x, mean, rstd);
  w_to_bf16_kernel<<<(NDIM * DDIM) / 1024, 256, 0, stream>>>(w, wbf);

  dim3 grid(DDIM / BD, NROWS / BM);             // (8, 256)
  ln_wmma_gemm_kernel<<<grid, 256, 0, stream>>>(x, wbf, bias, gamma, beta,
                                                mean, rstd, out);
}
